// CFGEncoder_73693048865004
// MI455X (gfx1250) — matrix operand order repaired and checked by reading, not compiled, not run on hardware
//
#include <hip/hip_runtime.h>

// ---------------------------------------------------------------------------
// CDNA5 (gfx1250) GraphSAGE: scatter-mean + fused split-bf16 WMMA GEMMs
// ---------------------------------------------------------------------------

typedef __attribute__((ext_vector_type(16))) __bf16 v16bf;
typedef __attribute__((ext_vector_type(8)))  float  v8f;

#define HIDN 256

// --------------------------- utility kernels -------------------------------

__global__ void k_zero(float* __restrict__ p, long long n) {
  long long i = (long long)blockIdx.x * blockDim.x + threadIdx.x;
  long long stride = (long long)gridDim.x * blockDim.x;
  for (; i < n; i += stride) p[i] = 0.0f;
}

__global__ void k_deg(const long long* __restrict__ dst, float* __restrict__ deg, int E) {
  int e = blockIdx.x * blockDim.x + threadIdx.x;
  if (e < E) atomicAdd(&deg[dst[e]], 1.0f);
}

__global__ void k_recip(float* __restrict__ deg, int n) {
  int i = blockIdx.x * blockDim.x + threadIdx.x;
  if (i < n) {
    float d = deg[i];
    deg[i] = (d > 0.0f) ? (1.0f / d) : 0.0f;
  }
}

// seg id + 1/count per node via binary search on CSR offsets (513 int64)
__global__ void k_seg(const long long* __restrict__ node_pos,
                      int* __restrict__ segn, float* __restrict__ scalen,
                      int n_nodes, int n_graphs) {
  int i = blockIdx.x * blockDim.x + threadIdx.x;
  if (i >= n_nodes) return;
  int lo = 0, hi = n_graphs;
  while (hi - lo > 1) {
    int mid = (lo + hi) >> 1;
    if ((long long)i >= node_pos[mid]) lo = mid; else hi = mid;
  }
  segn[i] = lo;
  scalen[i] = 1.0f / (float)(node_pos[lo + 1] - node_pos[lo]);
}

// per-(edge,feature) scatter-add: grid.x = E, block = dim
__global__ void k_scatter(const float* __restrict__ x,
                          const long long* __restrict__ src,
                          const long long* __restrict__ dst,
                          float* __restrict__ agg, int dim) {
  int e = blockIdx.x;
  int f = threadIdx.x;
  long long s = src[e], d = dst[e];
  atomicAdd(&agg[(size_t)d * dim + f], x[(size_t)s * dim + f]);
}

// Pack Wcat = [W0 (D0 rows); W1 (D1 rows)] (each row-major [k][256]) into
// WMMA B-fragment lane order as split bf16 (hi, lo).
// b[e] of lane l in tile (nT, ks): B[ks*32 + 8*(l>>4) + (e&7) + 16*(e>>3)][nT*16 + (l&15)]
// flat index: (((nT*KT + ks)*32 + lane)*16 + e)
__global__ void k_pack_w(const float* __restrict__ W0, int D0,
                         const float* __restrict__ W1, int D1,
                         __bf16* __restrict__ whi, __bf16* __restrict__ wlo) {
  int KT = (D0 + D1) >> 5;
  int total = 16 * KT * 32 * 16;
  int idx = blockIdx.x * blockDim.x + threadIdx.x;
  if (idx >= total) return;
  int e    = idx & 15;
  int lane = (idx >> 4) & 31;
  int ks   = (idx >> 9) % KT;
  int nT   = (idx >> 9) / KT;
  int n = nT * 16 + (lane & 15);
  int k = ks * 32 + 8 * (lane >> 4) + (e & 7) + ((e >> 3) << 4);
  float w = (k < D0) ? W0[(size_t)k * HIDN + n]
                     : W1[(size_t)(k - D0) * HIDN + n];
  __bf16 h = (__bf16)w;
  whi[idx] = h;
  wlo[idx] = (__bf16)(w - (float)h);
}

// ------------------------------ fused GEMM ---------------------------------
// C[m, 0:256] = (dinv[m]*A0[m,:] | A1[m,:]) @ Wcat   using split-bf16 WMMA.
// mode 0: out[m,n] = relu(C + bias[n])   (layer 1 -> h1)
// mode 1: atomicAdd(pooled[seg[m], n], C * scalen[m])   (layer 2 + pooling)
// Block: 256 threads = 8 waves; wave (w>>2) picks mTile within block's 32
// rows, (w&3) picks a 64-col quad. Grid.x = n_rows/32 (100000/32 = 3125).
__global__ void __launch_bounds__(256)
k_gemm_wmma(const float* __restrict__ A0, int D0,
            const float* __restrict__ A1, int D1,
            const float* __restrict__ dinv,
            const __bf16* __restrict__ Whi, const __bf16* __restrict__ Wlo,
            const float* __restrict__ bias,
            float* __restrict__ out,
            const int* __restrict__ segn, const float* __restrict__ scalen,
            float* __restrict__ pooled,
            int mode) {
  const int lane   = threadIdx.x & 31;
  const int wave   = threadIdx.x >> 5;
  const int mTile  = blockIdx.x * 2 + (wave >> 2);
  const int nQuad  = wave & 3;
  const int hiHalf = lane >> 4;           // 0: lanes 0-15, 1: lanes 16-31
  const int row    = mTile * 16 + (lane & 15);
  const int KT     = (D0 + D1) >> 5;
  const float dv   = dinv[row];

  v8f acc0 = {}, acc1 = {}, acc2 = {}, acc3 = {};

  for (int ks = 0; ks < KT; ++ks) {
    const int kb = ks * 32;
    // A fragment: 16 bf16/lane. ISA layout: VGPR0-3 hold K = (0..7)+8*hiHalf,
    // VGPR4-7 hold K = (16..23)+8*hiHalf. Each 32-wide K block lies entirely
    // inside A0 or A1 (both multiples of 32 wide).
    const float* base;
    float s;
    if (kb < D0) { base = A0 + (size_t)row * D0 + kb;        s = dv;   }
    else         { base = A1 + (size_t)row * D1 + (kb - D0); s = 1.0f; }
    const float4 c0a = *(const float4*)(base + 8 * hiHalf);
    const float4 c0b = *(const float4*)(base + 8 * hiHalf + 4);
    const float4 c1a = *(const float4*)(base + 16 + 8 * hiHalf);
    const float4 c1b = *(const float4*)(base + 16 + 8 * hiHalf + 4);

    float av[16];
    av[0]=c0a.x;  av[1]=c0a.y;  av[2]=c0a.z;  av[3]=c0a.w;
    av[4]=c0b.x;  av[5]=c0b.y;  av[6]=c0b.z;  av[7]=c0b.w;
    av[8]=c1a.x;  av[9]=c1a.y;  av[10]=c1a.z; av[11]=c1a.w;
    av[12]=c1b.x; av[13]=c1b.y; av[14]=c1b.z; av[15]=c1b.w;

    v16bf ahi, alo;
#pragma unroll
    for (int e = 0; e < 16; ++e) {
      float f = av[e] * s;
      __bf16 h = (__bf16)f;
      ahi[e] = h;
      alo[e] = (__bf16)(f - (float)h);
    }

    // 4 B tiles per wave; packed fragments are contiguous 32B per lane.
    const size_t bbase = (((size_t)(nQuad * 4) * KT + ks) * 32 + lane) * 16;
    const size_t bstep = (size_t)KT * 32 * 16;   // next nTile
#pragma unroll
    for (int t = 0; t < 4; ++t) {
      const size_t bo = bbase + (size_t)t * bstep;
      v16bf bh = *(const v16bf*)(Whi + bo);
      v16bf bl = *(const v16bf*)(Wlo + bo);
      v8f* acc = (t == 0) ? &acc0 : (t == 1) ? &acc1 : (t == 2) ? &acc2 : &acc3;
      // split-bf16: a*b ~= ahi*bhi + ahi*blo + alo*bhi  (drop lo*lo, ~2^-16)
      *acc = __builtin_amdgcn_wmma_f32_16x16x32_bf16(false, ahi, false, bh,
                                                     (short)0, *acc, false, false);
      *acc = __builtin_amdgcn_wmma_f32_16x16x32_bf16(false, ahi, false, bl,
                                                     (short)0, *acc, false, false);
      *acc = __builtin_amdgcn_wmma_f32_16x16x32_bf16(false, alo, false, bh,
                                                     (short)0, *acc, false, false);
    }
  }

  // Epilogue. C/D layout: VGPR v holds M = v + 8*hiHalf, col = lane&15.
  const int col0 = nQuad * 64 + (lane & 15);
#pragma unroll
  for (int t = 0; t < 4; ++t) {
    const int coln = col0 + t * 16;
    v8f acc = (t == 0) ? acc0 : (t == 1) ? acc1 : (t == 2) ? acc2 : acc3;
#pragma unroll
    for (int v = 0; v < 8; ++v) {
      const int m = mTile * 16 + v + 8 * hiHalf;
      float val = acc[v];
      if (mode == 0) {
        val += bias[coln];
        out[(size_t)m * HIDN + coln] = fmaxf(val, 0.0f);
      } else {
        atomicAdd(&pooled[(size_t)segn[m] * HIDN + coln], val * scalen[m]);
      }
    }
  }
}

__global__ void k_bias_out(float* __restrict__ outp, const float* __restrict__ b2, int n) {
  int i = blockIdx.x * blockDim.x + threadIdx.x;
  if (i < n) outp[i] += b2[i & (HIDN - 1)];
}

// ------------------------------- launcher ----------------------------------

extern "C" void kernel_launch(void* const* d_in, const int* in_sizes, int n_in,
                              void* d_out, int out_size, void* d_ws, size_t ws_size,
                              hipStream_t stream) {
  const float*     x    = (const float*)d_in[0];      // [N,128] f32
  const long long* ei   = (const long long*)d_in[1];  // [2,E] i64
  const long long* npos = (const long long*)d_in[2];  // [G+1] i64
  const float*     W1l  = (const float*)d_in[3];      // [128,256]
  const float*     b1   = (const float*)d_in[4];      // [256]
  const float*     W1r  = (const float*)d_in[5];      // [128,256]
  const float*     W2l  = (const float*)d_in[6];      // [256,256]
  const float*     b2   = (const float*)d_in[7];      // [256]
  const float*     W2r  = (const float*)d_in[8];      // [256,256]

  const int IN_DIM  = 128;
  const int N       = in_sizes[0] / IN_DIM;           // 100000
  const int E       = in_sizes[1] / 2;                // 1600000
  const int G       = in_sizes[2] - 1;                // 512

  const long long* srcp = ei;
  const long long* dstp = ei + E;

  // ---- workspace carve-up (256B aligned) ----
  char* ws = (char*)d_ws;
  size_t off = 0;
  auto carve = [&](size_t bytes) -> char* {
    char* p = ws + off;
    off = (off + bytes + 255) & ~(size_t)255;
    return p;
  };
  float*  deg    = (float*)carve((size_t)N * 4);            // deg -> deg_inv
  int*    segn   = (int*)  carve((size_t)N * 4);
  float*  scalen = (float*)carve((size_t)N * 4);
  float*  agg    = (float*)carve((size_t)N * HIDN * 4);     // agg1 (front half) / agg2
  float*  h1     = (float*)carve((size_t)N * HIDN * 4);
  __bf16* whi1   = (__bf16*)carve((size_t)256 * 256 * 2);
  __bf16* wlo1   = (__bf16*)carve((size_t)256 * 256 * 2);
  __bf16* whi2   = (__bf16*)carve((size_t)512 * 256 * 2);
  __bf16* wlo2   = (__bf16*)carve((size_t)512 * 256 * 2);
  (void)ws_size; (void)n_in;

  float* outp = (float*)d_out;

  // ---- degree & graph metadata ----
  k_zero<<<2048, 256, 0, stream>>>(deg, (long long)N);
  k_deg<<<(E + 255) / 256, 256, 0, stream>>>(dstp, deg, E);
  k_recip<<<(N + 255) / 256, 256, 0, stream>>>(deg, N);
  k_seg<<<(N + 255) / 256, 256, 0, stream>>>(npos, segn, scalen, N, G);

  // ---- weight packing (bf16 hi/lo, WMMA B-fragment order) ----
  {
    int tot1 = 16 * ((IN_DIM + IN_DIM) >> 5) * 512;   // KT=8
    int tot2 = 16 * ((HIDN + HIDN) >> 5) * 512;       // KT=16
    k_pack_w<<<(tot1 + 255) / 256, 256, 0, stream>>>(W1l, IN_DIM, W1r, IN_DIM, whi1, wlo1);
    k_pack_w<<<(tot2 + 255) / 256, 256, 0, stream>>>(W2l, HIDN,   W2r, HIDN,   whi2, wlo2);
  }

  // ---- layer 1: scatter + GEMM (relu, +b1) -> h1 ----
  k_zero<<<4096, 256, 0, stream>>>(agg, (long long)N * IN_DIM);
  k_scatter<<<E, IN_DIM, 0, stream>>>(x, srcp, dstp, agg, IN_DIM);
  k_gemm_wmma<<<N / 32, 256, 0, stream>>>(agg, IN_DIM, x, IN_DIM, deg,
                                          whi1, wlo1, b1, h1,
                                          nullptr, nullptr, nullptr, 0);

  // ---- layer 2: scatter + GEMM with fused graph-mean pooling ----
  k_zero<<<4096, 256, 0, stream>>>(agg, (long long)N * HIDN);
  k_scatter<<<E, HIDN, 0, stream>>>(h1, srcp, dstp, agg, HIDN);
  k_zero<<<(G * HIDN + 255) / 256, 256, 0, stream>>>(outp, (long long)G * HIDN);
  k_gemm_wmma<<<N / 32, 256, 0, stream>>>(agg, HIDN, h1, HIDN, deg,
                                          whi2, wlo2, nullptr, nullptr,
                                          segn, scalen, outp, 1);
  k_bias_out<<<(G * HIDN + 255) / 256, 256, 0, stream>>>(outp, b2, G * HIDN);
}
